// network_6811818132045
// MI455X (gfx1250) — matrix operand order repaired and checked by reading, not compiled, not run on hardware
//
#include <hip/hip_runtime.h>
#include <hip/hip_bf16.h>

// ---------------------------------------------------------------------------
// SNN recurrence (RLeaky + Leaky readout), T=16384 steps, H=512 neurons.
// Latency-bound sequential scan -> single persistent workgroup (1 WGP,
// 16 wave32s, 512 threads = 1 thread per neuron).
//
// Key idea: W_rec is constant across all 16384 steps, so pin ALL of it
// on-WGP: each wave owns 2 row-tiles (32 rows); row-tile 2w is held in
// 128 VGPRs (bf16 WMMA-A layout), row-tile 2w+1 in LDS (256 KB total).
// Per step each wave runs two independent 16-deep WMMA chains
// (v_wmma_f32_16x16x32_bf16) with ZERO global traffic on the critical path.
// ---------------------------------------------------------------------------

#define T_STEPS 16384
#define HDIM    512
#define BETA    0.8f
#define THRV    1.0f

typedef __attribute__((ext_vector_type(16))) __bf16       v16bf;
typedef __attribute__((ext_vector_type(8)))  float        v8f;
typedef __attribute__((ext_vector_type(4)))  unsigned int u32x4;

union ATile { v16bf v; u32x4 q[2]; };

// ---------------------------------------------------------------------------
// One-shot: rearrange W_rec [512,512] fp32 row-major into bf16 WMMA-A tiles.
// Tile (r in 0..31 row-tiles, t in 0..15 k-tiles) = 16x32 bf16, stored so
// that lane l reads its 16 elements (32B) contiguously:
//   flat = ((r*16 + t)*32 + lane)*16 + i
// A-matrix 16-bit layout (ISA 7.12.2):
//   lanes 0-15 : M=lane,    elems 0-7 -> K 0-7,  elems 8-15 -> K 16-23
//   lanes 16-31: M=lane-16, elems 0-7 -> K 8-15, elems 8-15 -> K 24-31
// ---------------------------------------------------------------------------
__global__ __launch_bounds__(256) void prep_wmma_layout(
    const float* __restrict__ W, __bf16* __restrict__ Wp)
{
    int idx  = blockIdx.x * 256 + threadIdx.x;   // 32*16*32*16 = 262144
    int i    = idx & 15;
    int lane = (idx >> 4) & 31;
    int t    = (idx >> 9) & 15;
    int r    = idx >> 13;
    int m    = lane & 15;
    int k    = (lane < 16) ? ((i < 8) ? i : i + 8)
                           : ((i < 8) ? i + 8 : i + 16);
    Wp[idx] = (__bf16)W[(r * 16 + m) * HDIM + t * 32 + k];
}

// ---------------------------------------------------------------------------
// Persistent recurrence kernel: 1 block, 512 threads = 16 wave32s.
// ---------------------------------------------------------------------------
__global__ __launch_bounds__(512) void snn_persistent(
    const float*  __restrict__ x,      // [T]
    const float*  __restrict__ W_in,   // [H]
    const float*  __restrict__ b_in,   // [H]
    const float*  __restrict__ b_rec,  // [H]
    const float*  __restrict__ W_out,  // [H]
    const float*  __restrict__ b_out,  // [1]
    const __bf16* __restrict__ Wp,     // [32][16][512] bf16 WMMA tiles
    float*        __restrict__ out)    // mem_out[T] | spk_hist[T*H] | mem_hist[T*H]
{
    // 16 waves x 16 k-tiles x 1KB = 256KB of LDS-resident W (odd row-tiles)
    __shared__ __align__(64) __bf16 Wlds[16 * 16 * 512];
    __shared__ __align__(64) __bf16 spk_lds[HDIM];  // spike vector (0/1, exact in bf16)
    __shared__ float y_lds[HDIM];                   // W_rec @ spk result
    __shared__ float red_lds[16];                   // per-wave readout partials

    const int tid = threadIdx.x;     // 0..511 == neuron id
    const int w   = tid >> 5;        // wave id 0..15
    const int l   = tid & 31;        // lane id

    // Per-neuron persistent state + constants (every thread owns a neuron).
    float mem1 = 0.f, mem2 = 0.f;
    const float win  = W_in[tid];
    const float bin  = b_in[tid];
    const float brec = b_rec[tid];
    const float wout = W_out[tid];
    const float bout = b_out[0];
    spk_lds[tid] = (__bf16)0.f;

    float* mem_out  = out;
    float* spk_hist = out + T_STEPS;
    float* mem_hist = out + T_STEPS + (size_t)T_STEPS * HDIM;

    // ---- One-shot fill: row-tile 2w -> registers, row-tile 2w+1 -> LDS ----
    ATile areg[16];
    {
        const __bf16* src0 = Wp + ((size_t)(2 * w) * 16) * 512 + l * 16;     // regs
        const __bf16* src1 = Wp + ((size_t)(2 * w + 1) * 16) * 512 + l * 16; // LDS
        __bf16* dst1 = Wlds + (w * 16) * 512 + l * 16;
        #pragma unroll
        for (int kt = 0; kt < 16; ++kt) {
            const u32x4* ap = (const u32x4*)(src0 + kt * 512);
            areg[kt].q[0] = ap[0];
            areg[kt].q[1] = ap[1];
            const u32x4* bp = (const u32x4*)(src1 + kt * 512);
            u32x4* dp = (u32x4*)(dst1 + kt * 512);
            dp[0] = bp[0];
            dp[1] = bp[1];
        }
    }
    const __bf16* wl = Wlds + (w * 16) * 512 + l * 16;
    const int bbase = (l & 16) >> 1;   // B operand: lanes 0-15 -> K 0-7,16-23; 16-31 -> K 8-15,24-31

    __syncthreads();

    for (int t = 0; t < T_STEPS; ++t) {
        // ---- Phase A: y = W_rec @ spk  (all waves, EXEC all-ones) ----
        // Two independent accumulation chains per wave: rows 32w..32w+15
        // from register-resident A, rows 32w+16..32w+31 from LDS-resident A.
        v8f acc0 = {};
        v8f acc1 = {};
        #pragma unroll
        for (int kt = 0; kt < 16; ++kt) {
            ATile b;   // B[k][n] = spk[32*kt + k] for all n (replicated columns)
            const u32x4* bp = (const u32x4*)(spk_lds + kt * 32 + bbase);
            b.q[0] = bp[0];
            b.q[1] = bp[2];
            acc0 = __builtin_amdgcn_wmma_f32_16x16x32_bf16(
                false, areg[kt].v, false, b.v, (short)0, acc0, false, false);
            ATile a1;
            const u32x4* lp = (const u32x4*)(wl + kt * 512);
            a1.q[0] = lp[0];
            a1.q[1] = lp[1];
            acc1 = __builtin_amdgcn_wmma_f32_16x16x32_bf16(
                false, a1.v, false, b.v, (short)0, acc1, false, false);
        }
        // D layout: VGPR g holds M=g (lanes 0-15) / M=g+8 (lanes 16-31);
        // columns are replicas, so lanes 0 and 16 (N=0) deposit 8 rows each.
        if ((l & 15) == 0) {
            const int mb = w * 32 + ((l >> 4) << 3);
            #pragma unroll
            for (int g = 0; g < 8; ++g) y_lds[mb + g]      = acc0[g];
            #pragma unroll
            for (int g = 0; g < 8; ++g) y_lds[mb + 16 + g] = acc1[g];
        }
        __syncthreads();

        // ---- Phase B: neuron update (every thread; neuron j = tid) ----
        {
            const float xt   = x[t];
            const float base = BETA * mem1 + xt * win + bin + y_lds[tid] + brec;
            const float rst  = (mem1 - THRV > 0.f) ? 1.f : 0.f;  // from PREVIOUS mem1
            const float m1n  = base - rst * base;                // reset_mechanism='zero'
            const float s    = (m1n - THRV > 0.f) ? 1.f : 0.f;   // spike
            mem2 = BETA * mem2 + s;                              // readout leak, no reset
            mem1 = m1n;
            spk_lds[tid] = (__bf16)s;                            // feeds next step's WMMA B

            const size_t o = (size_t)t * HDIM + tid;
            spk_hist[o] = s;
            mem_hist[o] = m1n;

            // Readout partial: sum_j W_out[j]*mem2[j] -- wave32 butterfly
            float rsum = wout * mem2;
            #pragma unroll
            for (int m = 16; m >= 1; m >>= 1)
                rsum += __shfl_xor(rsum, m, 32);
            if (l == 0) red_lds[w] = rsum;

            if (tid == 0 && t + 8 < T_STEPS)
                __builtin_prefetch(x + t + 8, 0, 0);             // global_prefetch_b8
        }
        __syncthreads();

        // ---- Phase C: reduce 16 partials -> mem_out[t] (wave 0) ----
        if (w == 0) {
            float v = (l < 16) ? red_lds[l] : 0.f;
            #pragma unroll
            for (int m = 8; m >= 1; m >>= 1)
                v += __shfl_xor(v, m, 32);
            if (l == 0) mem_out[t] = v + bout;
        }
        // No barrier needed here: next-step writers/readers of spk_lds,
        // red_lds and y_lds are all fenced by the two barriers above.
    }
}

extern "C" void kernel_launch(void* const* d_in, const int* in_sizes, int n_in,
                              void* d_out, int out_size, void* d_ws, size_t ws_size,
                              hipStream_t stream) {
    (void)in_sizes; (void)n_in; (void)out_size; (void)ws_size;
    const float* x     = (const float*)d_in[0];
    const float* W_in  = (const float*)d_in[1];
    const float* b_in  = (const float*)d_in[2];
    const float* W_rec = (const float*)d_in[3];
    const float* b_rec = (const float*)d_in[4];
    const float* W_out = (const float*)d_in[5];
    const float* b_out = (const float*)d_in[6];

    __bf16* Wp = (__bf16*)d_ws;   // 512 KB bf16 WMMA-tiled W_rec

    prep_wmma_layout<<<1024, 256, 0, stream>>>(W_rec, Wp);
    snn_persistent<<<1, 512, 0, stream>>>(x, W_in, b_in, b_rec, W_out, b_out,
                                          Wp, (float*)d_out);
}
